// VectorizedMACE_87892210745596
// MI455X (gfx1250) — matrix operand order repaired and checked by reading, not compiled, not run on hardware
//
#include <hip/hip_runtime.h>
#include <hip/hip_bf16.h>
#include <math.h>

// ---- problem constants (from reference) ----
#define NN   8192
#define EE   131072
#define ZZ   10
#define CC   64
#define HH   2
#define GG   8
#define NBF  8
#define SHC  16
#define HIDN 64
#define NL   2
#define RCUTF 5.0f

#define MSG_WAVES 4   // waves per block in k_edge_msg (share B tiles via LDS)

typedef __attribute__((ext_vector_type(16))) _Float16 v16h;
typedef __attribute__((ext_vector_type(8)))  float    v8f;

__device__ __forceinline__ float silu_f(float v) { return v / (1.0f + __expf(-v)); }

__device__ __forceinline__ v8f wmma_f16(v16h a, v16h b, v8f c) {
  // D = A(16x32 f16) * B(32x16 f16) + C(16x16 f32)
  return __builtin_amdgcn_wmma_f32_16x16x32_f16(false, a, false, b, (short)0, c,
                                                false, false);
}

// A fragment (16x32 tile, row-major source). Lane L: m=L&15, kbase=(L>>4)*16.
// v16h elem i: K=kbase+i.
__device__ __forceinline__ v16h afrag_f32(const float* base, int ld, int k0, int lane) {
  int m  = lane & 15;
  int kb = (lane >> 4) << 4;
  const float* p = base + (size_t)m * ld + k0;
  v16h a;
#pragma unroll
  for (int j = 0; j < 8; ++j) a[j] = (_Float16)p[kb + j];
#pragma unroll
  for (int j = 0; j < 8; ++j) a[8 + j] = (_Float16)p[kb + 8 + j];
  return a;
}
__device__ __forceinline__ v16h afrag_f16(const _Float16* base, int ld, int k0, int lane) {
  int m  = lane & 15;
  int kb = (lane >> 4) << 4;
  const _Float16* p = base + (size_t)m * ld + k0;
  v16h a;
#pragma unroll
  for (int j = 0; j < 8; ++j) a[j] = p[kb + j];
#pragma unroll
  for (int j = 0; j < 8; ++j) a[8 + j] = p[kb + 8 + j];
  return a;
}
// B fragment (32x16 tile) from K-contiguous ("[col][k] transposed") f16 weights.
// Lane L: col=L&15, kbase=(L>>4)*16, 16 contiguous K halves => two b128 loads.
__device__ __forceinline__ v16h bfrag_tr(const _Float16* bt, int ldk, int col0, int k0, int lane) {
  int n  = lane & 15;
  int kb = (lane >> 4) << 4;
  const _Float16* p = bt + (size_t)(col0 + n) * ldk + k0 + kb;
  v16h b;
#pragma unroll
  for (int j = 0; j < 16; ++j) b[j] = p[j];
  return b;
}

// ---------------- prep kernels ----------------
__global__ void k_species(const float* __restrict__ attrs, const int* __restrict__ batch,
                          const int* __restrict__ head, const float* __restrict__ E0,
                          int* spec, int* nhead, float* node_energy, float* contrib) {
  int n = blockIdx.x * blockDim.x + threadIdx.x;
  if (n >= NN) return;
  int s = 0; float best = -1.f;
#pragma unroll
  for (int z = 0; z < ZZ; ++z) { float v = attrs[n * ZZ + z]; if (v > best) { best = v; s = z; } }
  int g = batch[n];
  int h = head[g];
  spec[n] = s; nhead[n] = h;
  float e0 = E0[s * HH + h];
  node_energy[n] = e0;
  atomicAdd(&contrib[g * (2 + NL) + 0], e0);
}

__global__ void k_embed(const float* __restrict__ W_emb, const int* __restrict__ spec,
                        float* F0) {
  int idx = blockIdx.x * blockDim.x + threadIdx.x;
  if (idx >= NN * CC) return;
  int n = idx >> 6, c = idx & 63;
  F0[idx] = W_emb[spec[n] * CC + c];
}

__global__ void k_t64(const float* __restrict__ W, _Float16* Wt) {  // [64,64] -> f16 [col][k]
  int idx = blockIdx.x * blockDim.x + threadIdx.x;
  if (idx >= 64 * 64) return;
  int r = idx >> 6, c = idx & 63;
  Wt[c * 64 + r] = (_Float16)W[r * 64 + c];
}

__global__ void k_tw3(const float* __restrict__ w3, _Float16* w3t) { // [64,1024] -> f16 [1024][64]
  int idx = blockIdx.x * blockDim.x + threadIdx.x;
  if (idx >= 64 * 1024) return;
  int r = idx >> 10, c = idx & 1023;
  w3t[(size_t)c * 64 + r] = (_Float16)w3[idx];
}

// ---------------- edge geometry: SH + radial basis ----------------
__global__ void k_edge_geom(const float* __restrict__ pos, const float* __restrict__ shifts,
                            const int* __restrict__ eidx, float* Y, float* RBF) {
  int e = blockIdx.x * blockDim.x + threadIdx.x;
  if (e >= EE) return;
  int s = eidx[e], r = eidx[EE + e];
  float vx = pos[r * 3 + 0] - pos[s * 3 + 0] + shifts[e * 3 + 0];
  float vy = pos[r * 3 + 1] - pos[s * 3 + 1] + shifts[e * 3 + 1];
  float vz = pos[r * 3 + 2] - pos[s * 3 + 2] + shifts[e * 3 + 2];
  float len = sqrtf(vx * vx + vy * vy + vz * vz);
  float il = 1.f / (len + 1e-9f);
  float x = vx * il, y = vy * il, z = vz * il;
  const float s3 = 1.7320508075688772f, s5 = 2.23606797749979f;
  const float s15 = 3.872983346207417f, s7 = 2.6457513110645906f;
  const float c35 = 2.091650066335189f, c105 = 10.246950765959598f, c21 = 1.6201851746019651f;
  float* yo = Y + (size_t)e * SHC;
  yo[0] = 1.f;
  yo[1] = s3 * x; yo[2] = s3 * y; yo[3] = s3 * z;
  yo[4] = s15 * x * y; yo[5] = s15 * y * z; yo[6] = 0.5f * s5 * (3.f * z * z - 1.f);
  yo[7] = s15 * x * z; yo[8] = 0.5f * s15 * (x * x - y * y);
  yo[9] = c35 * y * (3.f * x * x - y * y); yo[10] = c105 * x * y * z;
  yo[11] = c21 * y * (5.f * z * z - 1.f); yo[12] = 0.5f * s7 * z * (5.f * z * z - 3.f);
  yo[13] = c21 * x * (5.f * z * z - 1.f); yo[14] = 0.5f * c105 * z * (x * x - y * y);
  yo[15] = c35 * x * (x * x - 3.f * y * y);
  float u = fminf(fmaxf(len / RCUTF, 0.f), 1.f);
  float u2 = u * u, u3 = u2 * u, u6 = u3 * u3, u7 = u6 * u, u8 = u7 * u;
  float fcut = (len < RCUTF) ? (1.f - 28.f * u6 + 48.f * u7 - 21.f * u8) : 0.f;
  float pre = 0.6324555320336759f * fcut / (len + 1e-9f);
  float* ro = RBF + (size_t)e * NBF;
#pragma unroll
  for (int i = 1; i <= NBF; ++i)
    ro[i - 1] = pre * __sinf((float)i * 3.14159265358979323f * len / RCUTF);
}

// ---------------- generic [M,64]@[64,64] WMMA GEMM ----------------
__global__ __launch_bounds__(32) void k_gemm64(const float* __restrict__ A,
                                               const _Float16* __restrict__ Bt,
                                               float* D) {
  int tile = blockIdx.x, lane = threadIdx.x;
  const float* ab = A + (size_t)tile * 16 * 64;
  v16h a0 = afrag_f32(ab, 64, 0, lane);
  v16h a1 = afrag_f32(ab, 64, 32, lane);
  int cl = lane & 15, rbi = (lane >> 4) << 3;
#pragma unroll
  for (int t = 0; t < 4; ++t) {
    v16h b0 = bfrag_tr(Bt, 64, t * 16, 0, lane);
    v16h b1 = bfrag_tr(Bt, 64, t * 16, 32, lane);
    v8f acc = {};
    acc = wmma_f16(a0, b0, acc);
    acc = wmma_f16(a1, b1, acc);
    int col = t * 16 + cl;
#pragma unroll
    for (int r = 0; r < 8; ++r)
      D[(size_t)(tile * 16 + rbi + r) * 64 + col] = acc[r];
  }
}

// ---------------- radial MLP hidden layers (small) ----------------
__global__ __launch_bounds__(256) void k_radial_mlp(const float* __restrict__ RBF,
                                                    const float* __restrict__ w1,
                                                    const float* __restrict__ w2,
                                                    _Float16* H2) {
  __shared__ float h1s[4][HIDN];
  int le = threadIdx.x >> 6, j = threadIdx.x & 63;
  int e = blockIdx.x * 4 + le;
  const float* rb = RBF + (size_t)e * NBF;
  float acc = 0.f;
#pragma unroll
  for (int i = 0; i < NBF; ++i) acc += rb[i] * w1[i * HIDN + j];
  h1s[le][j] = silu_f(acc);
  __syncthreads();
  float acc2 = 0.f;
#pragma unroll
  for (int i = 0; i < HIDN; ++i) acc2 += h1s[le][i] * w2[i * HIDN + j];
  H2[(size_t)e * HIDN + j] = (_Float16)silu_f(acc2);
}

// ---- fused w3 GEMM + message scale + scatter into A[N,SH,C] (1/AVG_NEIGH folded) ----
// 4 waves/block, 16 edges per wave. Each 2KB B tile (16 cols x 64 K, f16) is staged
// cooperatively into double-buffered LDS once per block instead of 4x from L2.
__global__ __launch_bounds__(32 * MSG_WAVES) void k_edge_msg(
    const _Float16* __restrict__ H2, const _Float16* __restrict__ w3t,
    const float* __restrict__ X, const float* __restrict__ Y,
    const int* __restrict__ eidx, float* AT) {
  __shared__ __align__(16) _Float16 Bs[2][16 * 64];

  int lane = threadIdx.x & 31;
  int wv   = threadIdx.x >> 5;
  int tile = blockIdx.x * MSG_WAVES + wv;  // 16-edge tile owned by this wave

  const _Float16* ab = H2 + (size_t)tile * 16 * HIDN;
  v16h a0 = afrag_f16(ab, HIDN, 0, lane);
  v16h a1 = afrag_f16(ab, HIDN, 32, lane);
  int cl = lane & 15, rbi = (lane >> 4) << 3;
  int snd[8], rcv[8];
#pragma unroll
  for (int r = 0; r < 8; ++r) {
    int e = tile * 16 + rbi + r;
    snd[r] = eidx[e];
    rcv[r] = eidx[EE + e];
  }

  // cooperative B-tile stage: 128 threads x 16B (8 halves) = 2048B = one tile
  auto loadB = [&](int t, int buf) {
    const float4* src = (const float4*)(w3t + (size_t)t * 16 * 64 + threadIdx.x * 8);
    *(float4*)(&Bs[buf][threadIdx.x * 8]) = *src;
  };

  loadB(0, 0);
  for (int t = 0; t < 64; ++t) {
    __syncthreads();  // Bs[t&1] stores visible; prior reads of Bs[(t+1)&1] retired
    int buf = t & 1;
    // lane: col = cl, 16 contiguous K halves at kbase=(lane>>4)*16 (+32 for chunk 1)
    const _Float16* bp = &Bs[buf][(size_t)cl * 64 + ((lane >> 4) << 4)];
    v16h b0, b1;
#pragma unroll
    for (int j = 0; j < 16; ++j) b0[j] = bp[j];
#pragma unroll
    for (int j = 0; j < 16; ++j) b1[j] = bp[32 + j];

    if (t + 1 < 64) {
      __builtin_prefetch(w3t + (size_t)(t + 1) * 16 * 64, 0, 0);
      loadB(t + 1, (t + 1) & 1);
    }

    v8f acc = {};
    acc = wmma_f16(a0, b0, acc);
    acc = wmma_f16(a1, b1, acc);

    int col = t * 16 + cl;  // global col in [0,1024): col = c*16 + k
    int c = col >> 4, k = col & 15;
#pragma unroll
    for (int r = 0; r < 8; ++r) {
      int e = tile * 16 + rbi + r;
      float v = acc[r] * X[(size_t)snd[r] * CC + c] * Y[(size_t)e * SHC + k] * 0.0625f;
      atomicAdd(&AT[(size_t)rcv[r] * (SHC * CC) + k * CC + c], v);
    }
  }
}

// ---- W_lin GEMM over [(n,k),c] with fused invariant reduction -> INV[n,d] ----
__global__ __launch_bounds__(32) void k_lin_inv(const float* __restrict__ AT,
                                                const _Float16* __restrict__ WlinT,
                                                const float* __restrict__ w_corr,
                                                float* INV) {
  int n = blockIdx.x, lane = threadIdx.x;
  const float* ab = AT + (size_t)n * SHC * CC;   // 16 rows (k) x 64 (c)
  v16h a0 = afrag_f32(ab, CC, 0, lane);
  v16h a1 = afrag_f32(ab, CC, 32, lane);
  int cl = lane & 15;
#pragma unroll
  for (int t = 0; t < 4; ++t) {
    v16h b0 = bfrag_tr(WlinT, 64, t * 16, 0, lane);
    v16h b1 = bfrag_tr(WlinT, 64, t * 16, 32, lane);
    v8f acc = {};
    acc = wmma_f16(a0, b0, acc);
    acc = wmma_f16(a1, b1, acc);
    // rows m = rbi + r are the k index; col d = t*16 + cl
    float ssq = 0.f;
#pragma unroll
    for (int r = 0; r < 8; ++r) ssq += acc[r] * acc[r];
    ssq += __shfl_xor(ssq, 16, 32);  // combine lane halves: full sum over k=0..15
    if (lane < 16) {                 // lanes 0..15 hold k=0 in acc[0]
      int d = t * 16 + cl;
      INV[(size_t)n * CC + d] = acc[0] + w_corr[d] * ssq;
    }
  }
}

// ---- element-dependent skip connection ----
__global__ __launch_bounds__(64) void k_skip(const float* __restrict__ F,
                                             const int* __restrict__ spec,
                                             const float* __restrict__ W_skip,
                                             float* SC) {
  __shared__ float fr[CC];
  int n = blockIdx.x, d = threadIdx.x;
  fr[d] = F[(size_t)n * CC + d];
  __syncthreads();
  const float* Wk = W_skip + (size_t)spec[n] * CC * CC;
  float acc = 0.f;
#pragma unroll 8
  for (int c = 0; c < CC; ++c) acc += fr[c] * Wk[c * CC + d];
  SC[(size_t)n * CC + d] = acc;
}

// ---- W_prod GEMM + skip add, writes next feats and d_out feature slab ----
__global__ __launch_bounds__(32) void k_prod(const float* __restrict__ INV,
                                             const _Float16* __restrict__ WprodT,
                                             const float* __restrict__ SC,
                                             float* Fnext, float* out_nf, int layer) {
  int tile = blockIdx.x, lane = threadIdx.x;
  const float* ab = INV + (size_t)tile * 16 * CC;
  v16h a0 = afrag_f32(ab, CC, 0, lane);
  v16h a1 = afrag_f32(ab, CC, 32, lane);
  int cl = lane & 15, rbi = (lane >> 4) << 3;
#pragma unroll
  for (int t = 0; t < 4; ++t) {
    v16h b0 = bfrag_tr(WprodT, 64, t * 16, 0, lane);
    v16h b1 = bfrag_tr(WprodT, 64, t * 16, 32, lane);
    v8f acc = {};
    acc = wmma_f16(a0, b0, acc);
    acc = wmma_f16(a1, b1, acc);
    int col = t * 16 + cl;
#pragma unroll
    for (int r = 0; r < 8; ++r) {
      int row = tile * 16 + rbi + r;
      float v = acc[r] + SC[(size_t)row * CC + col];
      Fnext[(size_t)row * CC + col] = v;
      out_nf[(size_t)row * (NL * CC) + layer * CC + col] = v;
    }
  }
}

// ---- per-layer readout + energy accumulation ----
__global__ void k_readout(const float* __restrict__ F, const float* __restrict__ W_read,
                          const int* __restrict__ nhead, const int* __restrict__ batch,
                          float* node_energy, float* contrib, int layer) {
  int n = blockIdx.x * blockDim.x + threadIdx.x;
  if (n >= NN) return;
  int h = nhead[n];
  float acc = 0.f;
#pragma unroll 8
  for (int c = 0; c < CC; ++c) acc += F[(size_t)n * CC + c] * W_read[c * HH + h];
  node_energy[n] += acc;
  atomicAdd(&contrib[batch[n] * (2 + NL) + 2 + layer], acc);
}

__global__ void k_final(const float* __restrict__ contrib, float* total) {
  int g = threadIdx.x;
  if (g < GG)
    total[g] = contrib[g * 4] + contrib[g * 4 + 1] + contrib[g * 4 + 2] + contrib[g * 4 + 3];
}

// ================= host side =================
struct LayerP {
  const float *W_up, *w1, *w2, *w3, *W_lin, *W_skip, *w_corr, *W_prod, *W_read;
};

extern "C" void kernel_launch(void* const* d_in, const int* in_sizes, int n_in,
                              void* d_out, int out_size, void* d_ws, size_t ws_size,
                              hipStream_t stream) {
  const float *pos, *attrs, *shifts, *W_emb, *E0;
  const int *edge, *batch, *head;
  LayerP L[2];

  if (in_sizes[0] == NN * 3) {
    // dict-insertion order: pos, attrs, shifts, [params insertion], edge, batch, head, ptr
    pos = (const float*)d_in[0]; attrs = (const float*)d_in[1]; shifts = (const float*)d_in[2];
    W_emb = (const float*)d_in[3]; E0 = (const float*)d_in[4];
    for (int l = 0; l < 2; ++l) {
      int b = 5 + l * 9;
      L[l].W_up   = (const float*)d_in[b + 0];
      L[l].w1     = (const float*)d_in[b + 1];
      L[l].w2     = (const float*)d_in[b + 2];
      L[l].w3     = (const float*)d_in[b + 3];
      L[l].W_lin  = (const float*)d_in[b + 4];
      L[l].W_skip = (const float*)d_in[b + 5];
      L[l].w_corr = (const float*)d_in[b + 6];
      L[l].W_prod = (const float*)d_in[b + 7];
      L[l].W_read = (const float*)d_in[b + 8];
    }
    edge = (const int*)d_in[23]; batch = (const int*)d_in[24]; head = (const int*)d_in[25];
  } else {
    // pytree-sorted order: batch, edge, head, attrs, [params sorted], pos, ptr, shifts
    batch = (const int*)d_in[0]; edge = (const int*)d_in[1]; head = (const int*)d_in[2];
    attrs = (const float*)d_in[3];
    W_emb = (const float*)d_in[4]; E0 = (const float*)d_in[5];
    for (int l = 0; l < 2; ++l) {
      int b = 6 + l * 9;  // W_lin, W_prod, W_read, W_skip, W_up, w1, w2, w3, w_corr
      L[l].W_lin  = (const float*)d_in[b + 0];
      L[l].W_prod = (const float*)d_in[b + 1];
      L[l].W_read = (const float*)d_in[b + 2];
      L[l].W_skip = (const float*)d_in[b + 3];
      L[l].W_up   = (const float*)d_in[b + 4];
      L[l].w1     = (const float*)d_in[b + 5];
      L[l].w2     = (const float*)d_in[b + 6];
      L[l].w3     = (const float*)d_in[b + 7];
      L[l].w_corr = (const float*)d_in[b + 8];
    }
    pos = (const float*)d_in[24]; shifts = (const float*)d_in[26];
  }

  // ---- workspace carving ----
  char* wsb = (char*)d_ws;
  size_t off = 0;
  auto take = [&](size_t bytes) -> void* {
    void* p = wsb + off;
    off = (off + bytes + 255) & ~(size_t)255;
    return p;
  };
  float*    Y    = (float*)take((size_t)EE * SHC * 4);
  float*    RBF  = (float*)take((size_t)EE * NBF * 4);
  _Float16* H2   = (_Float16*)take((size_t)EE * HIDN * 2);
  float*    X    = (float*)take((size_t)NN * CC * 4);
  float*    F0   = (float*)take((size_t)NN * CC * 4);
  float*    F1   = (float*)take((size_t)NN * CC * 4);
  float*    AT   = (float*)take((size_t)NN * SHC * CC * 4);
  float*    INV  = (float*)take((size_t)NN * CC * 4);
  float*    SC   = (float*)take((size_t)NN * CC * 4);
  _Float16* W3T[2]; _Float16* WUPT[2]; _Float16* WLINT[2]; _Float16* WPRODT[2];
  for (int l = 0; l < 2; ++l) {
    W3T[l]    = (_Float16*)take((size_t)1024 * 64 * 2);
    WUPT[l]   = (_Float16*)take((size_t)64 * 64 * 2);
    WLINT[l]  = (_Float16*)take((size_t)64 * 64 * 2);
    WPRODT[l] = (_Float16*)take((size_t)64 * 64 * 2);
  }
  int* SPEC  = (int*)take((size_t)NN * 4);
  int* NHEAD = (int*)take((size_t)NN * 4);

  // ---- output layout: [total(8) | node_energy(8192) | contrib(32) | node_feats(8192*128)] ----
  float* out         = (float*)d_out;
  float* total       = out;
  float* node_energy = out + GG;
  float* contrib     = out + GG + NN;
  float* out_nf      = out + GG + NN + GG * (2 + NL);

  hipMemsetAsync(d_out, 0, (size_t)(GG + NN + GG * (2 + NL)) * 4, stream);

  k_species<<<(NN + 255) / 256, 256, 0, stream>>>(attrs, batch, head, E0, SPEC, NHEAD,
                                                  node_energy, contrib);
  k_embed<<<(NN * CC + 255) / 256, 256, 0, stream>>>(W_emb, SPEC, F0);
  for (int l = 0; l < 2; ++l) {
    k_tw3<<<(64 * 1024 + 255) / 256, 256, 0, stream>>>(L[l].w3, W3T[l]);
    k_t64<<<16, 256, 0, stream>>>(L[l].W_up, WUPT[l]);
    k_t64<<<16, 256, 0, stream>>>(L[l].W_lin, WLINT[l]);
    k_t64<<<16, 256, 0, stream>>>(L[l].W_prod, WPRODT[l]);
  }
  k_edge_geom<<<EE / 256, 256, 0, stream>>>(pos, shifts, edge, Y, RBF);

  float* Fcur = F0;
  float* Fnext = F1;
  for (int l = 0; l < 2; ++l) {
    k_gemm64<<<NN / 16, 32, 0, stream>>>(Fcur, WUPT[l], X);               // x = feats @ W_up
    k_radial_mlp<<<EE / 4, 256, 0, stream>>>(RBF, L[l].w1, L[l].w2, H2);  // hidden layers
    hipMemsetAsync(AT, 0, (size_t)NN * SHC * CC * 4, stream);
    k_edge_msg<<<EE / (16 * MSG_WAVES), 32 * MSG_WAVES, 0, stream>>>(H2, W3T[l], X, Y,
                                                                     edge, AT);
    k_lin_inv<<<NN, 32, 0, stream>>>(AT, WLINT[l], L[l].w_corr, INV);     // W_lin + invariants
    k_skip<<<NN, 64, 0, stream>>>(Fcur, SPEC, L[l].W_skip, SC);           // skip connection
    k_prod<<<NN / 16, 32, 0, stream>>>(INV, WPRODT[l], SC, Fnext, out_nf, l);
    k_readout<<<NN / 256, 256, 0, stream>>>(Fnext, L[l].W_read, NHEAD, batch,
                                            node_energy, contrib, l);
    float* t = Fcur; Fcur = Fnext; Fnext = t;
  }
  k_final<<<1, 32, 0, stream>>>(contrib, total);
}
